// ImbaHgnn_41283225649262
// MI455X (gfx1250) — compile-verified, matrix operand order checked by reading, not run on hardware
//
#include <hip/hip_runtime.h>
#include <hip/hip_bf16.h>

typedef __attribute__((ext_vector_type(16))) __bf16 v16bf;
typedef __attribute__((ext_vector_type(8)))  __bf16 v8bf;
typedef __attribute__((ext_vector_type(8)))  float  v8f;

#define SLOPE_C 0.2f
#define NEG_INF_C -1000000000.0f

// 16-bit A/B fragment k-pattern (ISA 05_wmma, 16x32): per lane two contiguous
// 8-element runs starting at base and base+16, base = (lane>>4)*8.
// Fragment element i (0..15) <-> k = base + (i<8 ? i : i+8).

__device__ __forceinline__ float elu1(float v) { return v > 0.f ? v : __expf(v) - 1.f; }

#define PACK8(vec, off, q0, q1)                                                          \
  vec[(off)+0]=(__bf16)(q0).x; vec[(off)+1]=(__bf16)(q0).y; vec[(off)+2]=(__bf16)(q0).z; \
  vec[(off)+3]=(__bf16)(q0).w; vec[(off)+4]=(__bf16)(q1).x; vec[(off)+5]=(__bf16)(q1).y; \
  vec[(off)+6]=(__bf16)(q1).z; vec[(off)+7]=(__bf16)(q1).w;

// ---------------------------------------------------------------------------
// GEMM: C[M x Nc] = act(A[M x K] @ W[Nc x K]^T + bias), bf16 WMMA, f32 acc.
// REQUIRES K % 32 == 0 (all call sites: 64/256/512). Unconditional b128 loads.
// ACT: 0=none, 1=elu, 2=sigmoid. HAS_CT: also write bf16 transposed CT[Nc x M].
// One wave per 16x16 tile; interior tiles take a guard-free store path.
// ---------------------------------------------------------------------------
template <int ACT, bool HAS_CT, bool HAS_BIAS>
__global__ void gemm_bias_act(const float* __restrict__ A, int lda, int M,
                              const float* __restrict__ W, int ldw, int Nc,
                              const float* __restrict__ bias, int K,
                              float* __restrict__ C, int ldc,
                              __bf16* __restrict__ CT, int ldct,
                              int tilesN, int tilesTotal)
{
  int wid  = threadIdx.x >> 5;
  int lane = threadIdx.x & 31;
  int tile = blockIdx.x * 4 + wid;
  if (tile >= tilesTotal) return;            // wave-uniform exit (EXEC stays all-ones)
  int tm = tile / tilesN, tn = tile % tilesN;
  int m0 = tm * 16, n0 = tn * 16;
  int hi = lane >> 4, lo = lane & 15, base = hi * 8;
  int arow = m0 + lo; if (arow > M  - 1) arow = M  - 1;
  int brow = n0 + lo; if (brow > Nc - 1) brow = Nc - 1;
  const float* ap = A + (long)arow * lda;
  const float* wp = W + (long)brow * ldw;
  v8f acc = {0.f,0.f,0.f,0.f,0.f,0.f,0.f,0.f};
  for (int kb = 0; kb < K; kb += 32) {
    const float* apk = ap + kb + base;
    const float* wpk = wp + kb + base;
    float4 xa0 = *(const float4*)(apk);
    float4 xa1 = *(const float4*)(apk + 4);
    float4 xa2 = *(const float4*)(apk + 16);
    float4 xa3 = *(const float4*)(apk + 20);
    float4 xb0 = *(const float4*)(wpk);
    float4 xb1 = *(const float4*)(wpk + 4);
    float4 xb2 = *(const float4*)(wpk + 16);
    float4 xb3 = *(const float4*)(wpk + 20);
    v16bf a, b;
    PACK8(a, 0, xa0, xa1); PACK8(a, 8, xa2, xa3);
    PACK8(b, 0, xb0, xb1); PACK8(b, 8, xb2, xb3);
    acc = __builtin_amdgcn_wmma_f32_16x16x32_bf16(false, a, false, b, (short)0, acc, false, false);
  }
  const float bval = HAS_BIAS ? bias[brow] : 0.f;  // brow == n0+lo clamped (store-guarded)
  const bool interior = (m0 + 16 <= M) && (n0 + 16 <= Nc);
  if (interior) {
#pragma unroll
    for (int r = 0; r < 8; ++r) {
      int row = m0 + r + 8 * hi;   // C/D: VGPR r -> M = r + 8*(lane>>4), N = lane&15
      int col = n0 + lo;
      float v = acc[r] + bval;
      if (ACT == 1)      v = elu1(v);
      else if (ACT == 2) v = 1.f / (1.f + __expf(-v));
      C[(long)row * ldc + col] = v;
      if (HAS_CT) CT[(long)col * ldct + row] = (__bf16)v;
    }
  } else {
#pragma unroll
    for (int r = 0; r < 8; ++r) {
      int row = m0 + r + 8 * hi;
      int col = n0 + lo;
      if (row < M && col < Nc) {
        float v = acc[r] + bval;
        if (ACT == 1)      v = elu1(v);
        else if (ACT == 2) v = 1.f / (1.f + __expf(-v));
        C[(long)row * ldc + col] = v;
        if (HAS_CT) CT[(long)col * ldct + row] = (__bf16)v;
      }
    }
  }
}

// 4 p-values from vectorized inputs (hard adjacency mask)
__device__ __forceinline__ float4 p4_hard(float4 f2q, float4 aq, float f1m, float rmax, float& lsum) {
  float4 r;
  float lg;
  lg = f1m + f2q.x; lg = lg > 0.f ? lg : SLOPE_C * lg; r.x = __expf((aq.x > 0.f ? lg : NEG_INF_C) - rmax);
  lg = f1m + f2q.y; lg = lg > 0.f ? lg : SLOPE_C * lg; r.y = __expf((aq.y > 0.f ? lg : NEG_INF_C) - rmax);
  lg = f1m + f2q.z; lg = lg > 0.f ? lg : SLOPE_C * lg; r.z = __expf((aq.z > 0.f ? lg : NEG_INF_C) - rmax);
  lg = f1m + f2q.w; lg = lg > 0.f ? lg : SLOPE_C * lg; r.w = __expf((aq.w > 0.f ? lg : NEG_INF_C) - rmax);
  lsum += r.x + r.y + r.z + r.w;
  return r;
}
// 4 p-values, soft mode: e = leaky + (mask ? pred : realm)
__device__ __forceinline__ float4 p4_soft(float4 f2q, uchar4 mq, float4 pq, float4 rq,
                                          float f1m, float rmax, float& lsum) {
  float4 r;
  float lg;
  lg = f1m + f2q.x; lg = lg > 0.f ? lg : SLOPE_C * lg; r.x = __expf(lg + (mq.x ? pq.x : rq.x) - rmax);
  lg = f1m + f2q.y; lg = lg > 0.f ? lg : SLOPE_C * lg; r.y = __expf(lg + (mq.y ? pq.y : rq.y) - rmax);
  lg = f1m + f2q.z; lg = lg > 0.f ? lg : SLOPE_C * lg; r.z = __expf(lg + (mq.z ? pq.z : rq.z) - rmax);
  lg = f1m + f2q.w; lg = lg > 0.f ? lg : SLOPE_C * lg; r.w = __expf(lg + (mq.w ? pq.w : rq.w) - rmax);
  lsum += r.x + r.y + r.z + r.w;
  return r;
}

// ---------------------------------------------------------------------------
// Fused graph attention, out[i] = elu( softmax_j(e_ij) @ fts ).
// Block = 128 threads (4 waves) per 16 rows. Pass1: row max. Pass2: per chunk of
// 128 columns, wave w generates P sub-chunk [32w,32w+32) in fp32, packs bf16 into
// double-buffered LDS, then all waves run 4 WMMA steps (A from LDS, B = fts^T bf16
// b128 loads). Row sums partitioned across all 128 threads. Requires N % 8 == 0.
// ---------------------------------------------------------------------------
template <bool SOFT>
__global__ void attn_fused(const float* __restrict__ f1v, const float* __restrict__ f2v,
                           const float* __restrict__ adj,
                           const unsigned char* __restrict__ mask,
                           const float* __restrict__ pred,
                           const float* __restrict__ realm,
                           const __bf16* __restrict__ ftsT, int ldft,
                           int N,
                           float* __restrict__ outp, int ldo, int col_off)
{
  __shared__ float sred[128];
  __shared__ float smax[16];
  __shared__ float ssum[16];
  __shared__ __align__(16) __bf16 Pbuf[2][4][16][40];  // [buf][subchunk][row][32 + pad]

  const int t = threadIdx.x;
  const int wid = t >> 5, lane = t & 31;
  const int hi = lane >> 4, lo = lane & 15, base = hi * 8;
  const int m0 = blockIdx.x * 16;

  // ---- pass 1: row max (16 rows x 8 threads, float4 sweeps; N % 4 == 0) ----
  {
    int r1 = t >> 3, c1 = t & 7;
    int rowi = m0 + r1; if (rowi > N - 1) rowi = N - 1;
    const float f1r = f1v[rowi];
    const long rb = (long)rowi * N;
    float mx = -3.0e38f;
    for (int j = c1 * 4; j < N; j += 32) {
      float4 f2q = *(const float4*)(f2v + j);
      float lg[4] = {f1r + f2q.x, f1r + f2q.y, f1r + f2q.z, f1r + f2q.w};
#pragma unroll
      for (int q = 0; q < 4; ++q) lg[q] = lg[q] > 0.f ? lg[q] : SLOPE_C * lg[q];
      if (SOFT) {
        uchar4 mq = *(const uchar4*)(mask + rb + j);
        float4 pq = *(const float4*)(pred + rb + j);
        float4 rq = *(const float4*)(realm + rb + j);
        if (j + 1024 < N) __builtin_prefetch(pred + rb + j + 1024, 0, 1);
        float e0 = lg[0] + (mq.x ? pq.x : rq.x);
        float e1 = lg[1] + (mq.y ? pq.y : rq.y);
        float e2 = lg[2] + (mq.z ? pq.z : rq.z);
        float e3 = lg[3] + (mq.w ? pq.w : rq.w);
        mx = fmaxf(mx, fmaxf(fmaxf(e0, e1), fmaxf(e2, e3)));
      } else {
        float4 aq = *(const float4*)(adj + rb + j);
        if (j + 1024 < N) __builtin_prefetch(adj + rb + j + 1024, 0, 1);
        float e0 = aq.x > 0.f ? lg[0] : NEG_INF_C;
        float e1 = aq.y > 0.f ? lg[1] : NEG_INF_C;
        float e2 = aq.z > 0.f ? lg[2] : NEG_INF_C;
        float e3 = aq.w > 0.f ? lg[3] : NEG_INF_C;
        mx = fmaxf(mx, fmaxf(fmaxf(e0, e1), fmaxf(e2, e3)));
      }
    }
    sred[t] = mx;
  }
  __syncthreads();
  if (t < 16) {
    float mx = sred[t * 8];
#pragma unroll
    for (int q = 1; q < 8; ++q) mx = fmaxf(mx, sred[t * 8 + q]);
    smax[t] = mx;
  }
  __syncthreads();

  // ---- pass 2 ----
  const int prow = (m0 + lo > N - 1) ? N - 1 : (m0 + lo);  // A layout: lane row = lane&15
  const float f1m  = f1v[prow];
  const float rmax = smax[lo];
  const long  prb  = (long)prow * N;
  const __bf16* frBase = ftsT + (long)(wid * 16 + lo) * ldft;  // B col = lane&15
  float lsum = 0.f;
  v8f acc = {0.f,0.f,0.f,0.f,0.f,0.f,0.f,0.f};
  const int nchunks = (N + 127) / 128;
  int bufi = 0;
  for (int cc = 0; cc < nchunks; ++cc, bufi ^= 1) {
    const int k0 = cc * 128;
    const int kw = k0 + wid * 32 + base;     // this lane's first run start
    const bool full = (k0 + 128 <= N);       // uniform across block
    v8bf run1, run2;
    if (full) {
      float4 f2a = *(const float4*)(f2v + kw);
      float4 f2b = *(const float4*)(f2v + kw + 4);
      float4 f2c = *(const float4*)(f2v + kw + 16);
      float4 f2d = *(const float4*)(f2v + kw + 20);
      float4 p0, p1, p2, p3;
      if (SOFT) {
        uchar4 m0q = *(const uchar4*)(mask + prb + kw);
        uchar4 m1q = *(const uchar4*)(mask + prb + kw + 4);
        uchar4 m2q = *(const uchar4*)(mask + prb + kw + 16);
        uchar4 m3q = *(const uchar4*)(mask + prb + kw + 20);
        float4 pa = *(const float4*)(pred + prb + kw);
        float4 pb = *(const float4*)(pred + prb + kw + 4);
        float4 pc = *(const float4*)(pred + prb + kw + 16);
        float4 pd = *(const float4*)(pred + prb + kw + 20);
        float4 ra = *(const float4*)(realm + prb + kw);
        float4 rb2 = *(const float4*)(realm + prb + kw + 4);
        float4 rc = *(const float4*)(realm + prb + kw + 16);
        float4 rd = *(const float4*)(realm + prb + kw + 20);
        p0 = p4_soft(f2a, m0q, pa, ra, f1m, rmax, lsum);
        p1 = p4_soft(f2b, m1q, pb, rb2, f1m, rmax, lsum);
        p2 = p4_soft(f2c, m2q, pc, rc, f1m, rmax, lsum);
        p3 = p4_soft(f2d, m3q, pd, rd, f1m, rmax, lsum);
      } else {
        float4 a0 = *(const float4*)(adj + prb + kw);
        float4 a1 = *(const float4*)(adj + prb + kw + 4);
        float4 a2 = *(const float4*)(adj + prb + kw + 16);
        float4 a3 = *(const float4*)(adj + prb + kw + 20);
        p0 = p4_hard(f2a, a0, f1m, rmax, lsum);
        p1 = p4_hard(f2b, a1, f1m, rmax, lsum);
        p2 = p4_hard(f2c, a2, f1m, rmax, lsum);
        p3 = p4_hard(f2d, a3, f1m, rmax, lsum);
      }
      PACK8(run1, 0, p0, p1);
      PACK8(run2, 0, p2, p3);
    } else {
      // tail chunk: branch-free clamped-scalar path
#pragma unroll
      for (int i = 0; i < 16; ++i) {
        int kk = kw + ((i < 8) ? i : (i + 8));
        int kc = (kk < N) ? kk : 0;
        float lg = f1m + f2v[kc];
        lg = lg > 0.f ? lg : SLOPE_C * lg;
        float ev;
        if (SOFT) {
          long idx = prb + kc;
          ev = lg + (mask[idx] ? pred[idx] : realm[idx]);
        } else {
          ev = (adj[prb + kc] > 0.f) ? lg : NEG_INF_C;
        }
        float p = (kk < N) ? __expf(ev - rmax) : 0.f;
        lsum += p;
        if (i < 8) run1[i] = (__bf16)p; else run2[i - 8] = (__bf16)p;
      }
    }
    *(v8bf*)&Pbuf[bufi][wid][lo][base]      = run1;
    *(v8bf*)&Pbuf[bufi][wid][lo][base + 16] = run2;
    __syncthreads();
#pragma unroll
    for (int c2 = 0; c2 < 4; ++c2) {
      v8bf a0 = *(const v8bf*)&Pbuf[bufi][c2][lo][base];
      v8bf a1 = *(const v8bf*)&Pbuf[bufi][c2][lo][base + 16];
      v16bf a = __builtin_shufflevector(a0, a1, 0,1,2,3,4,5,6,7,8,9,10,11,12,13,14,15);
      int kb2 = k0 + c2 * 32 + base;
      v16bf b;
      if (full) {
        v8bf b0 = *(const v8bf*)(frBase + kb2);
        v8bf b1 = *(const v8bf*)(frBase + kb2 + 16);
        b = __builtin_shufflevector(b0, b1, 0,1,2,3,4,5,6,7,8,9,10,11,12,13,14,15);
      } else {
#pragma unroll
        for (int i = 0; i < 16; ++i) {
          int kk = kb2 + ((i < 8) ? i : (i + 8));
          int kc = (kk < N) ? kk : 0;   // a==0 there, any finite b ok
          b[i] = frBase[kc];
        }
      }
      acc = __builtin_amdgcn_wmma_f32_16x16x32_bf16(false, a, false, b, (short)0, acc, false, false);
    }
  }
  sred[t] = lsum;                       // unique partition over all 128 threads
  __syncthreads();
  if (t < 16) {
    float s = 0.f;
#pragma unroll
    for (int q = 0; q < 8; ++q) s += sred[t + q * 16];
    ssum[t] = s;
  }
  __syncthreads();
  if (m0 + 16 <= N) {
#pragma unroll
    for (int r = 0; r < 8; ++r) {
      int rr = r + 8 * hi;
      float v = elu1(acc[r] / ssum[rr]);
      outp[(long)(m0 + rr) * ldo + col_off + wid * 16 + lo] = v;
    }
  } else {
#pragma unroll
    for (int r = 0; r < 8; ++r) {
      int rr = r + 8 * hi;
      int row = m0 + rr;
      if (row < N) {
        float v = elu1(acc[r] / ssum[rr]);
        outp[(long)row * ldo + col_off + wid * 16 + lo] = v;
      }
    }
  }
}

// f1[i] = fts[i,:] . a1 + a1b ; f2[i] = fts[i,:] . a2 + a2b  (row width 64)
__global__ void f1f2_kernel(const float* __restrict__ fts,
                            const float* __restrict__ a1, const float* __restrict__ a1b,
                            const float* __restrict__ a2, const float* __restrict__ a2b,
                            float* __restrict__ f1, float* __restrict__ f2, int N)
{
  int i = blockIdx.x * blockDim.x + threadIdx.x;
  if (i >= N) return;
  const float4* row = (const float4*)(fts + (long)i * 64);
  const float4* w1  = (const float4*)a1;
  const float4* w2  = (const float4*)a2;
  float s1 = 0.f, s2 = 0.f;
#pragma unroll
  for (int q = 0; q < 16; ++q) {
    float4 v = row[q], x = w1[q], y = w2[q];
    s1 += v.x*x.x + v.y*x.y + v.z*x.z + v.w*x.w;
    s2 += v.x*y.x + v.y*y.y + v.z*y.z + v.w*y.w;
  }
  f1[i] = s1 + a1b[0];
  f2[i] = s2 + a2b[0];
}

__global__ void gather64(const float* __restrict__ src, const int* __restrict__ idx,
                         float* __restrict__ dst, int nrows)
{
  int g = blockIdx.x * 256 + threadIdx.x;
  if (g >= nrows * 64) return;
  int r = g >> 6, c = g & 63;
  dst[g] = src[(long)idx[r] * 64 + c];
}

__global__ void combine3(const float* __restrict__ a, const float* __restrict__ b,
                         const float* __restrict__ c,
                         const float* __restrict__ rw, const float* __restrict__ rb,
                         float* __restrict__ outv, int n)
{
  int i = blockIdx.x * 256 + threadIdx.x;
  if (i >= n) return;
  float v = a[i] * rw[0] + b[i] * rw[1] + c[i] * rw[2] + rb[0];
  outv[i] = elu1(v);
}

__global__ void loss_partial(const float* __restrict__ h1, const float* __restrict__ h1p,
                             const int* __restrict__ idx_tir, const int* __restrict__ idx_rit,
                             float* __restrict__ partial, int nelem)
{
  __shared__ float red[256];
  int t = threadIdx.x;
  float s = 0.f;
  for (int e = blockIdx.x * 256 + t; e < nelem; e += gridDim.x * 256) {
    int r = e >> 6, c = e & 63;
    float d = h1p[(long)idx_rit[r] * 64 + c] - h1[(long)idx_tir[r] * 64 + c];
    s += d * d;
  }
  red[t] = s; __syncthreads();
  for (int off = 128; off > 0; off >>= 1) { if (t < off) red[t] += red[t + off]; __syncthreads(); }
  if (t == 0) partial[blockIdx.x] = red[0];
}

__global__ void loss_finish(const float* __restrict__ partial, int n,
                            float* __restrict__ outv, float scale)
{
  __shared__ float red[256];
  int t = threadIdx.x;
  float s = 0.f;
  for (int i = t; i < n; i += 256) s += partial[i];
  red[t] = s; __syncthreads();
  for (int off = 128; off > 0; off >>= 1) { if (t < off) red[t] += red[t + off]; __syncthreads(); }
  if (t == 0) outv[0] = red[0] * scale;
}

__global__ void fc2_logsoftmax(const float* __restrict__ o,
                               const float* __restrict__ W, const float* __restrict__ b,
                               float* __restrict__ outv, int M)
{
  int row = blockIdx.x * 256 + threadIdx.x;
  if (row >= M) return;
  const float* orow = o + (long)row * 45;
  float c[5];
#pragma unroll
  for (int j = 0; j < 5; ++j) {
    float s = b[j];
    for (int k = 0; k < 45; ++k) s += orow[k] * W[j * 45 + k];
    c[j] = s;
  }
  float m = c[0];
#pragma unroll
  for (int j = 1; j < 5; ++j) m = fmaxf(m, c[j]);
  float sum = 0.f;
#pragma unroll
  for (int j = 0; j < 5; ++j) sum += __expf(c[j] - m);
  float lse = m + __logf(sum);
#pragma unroll
  for (int j = 0; j < 5; ++j) outv[(long)row * 5 + j] = c[j] - lse;
}

// ---------------------------------------------------------------------------
extern "C" void kernel_launch(void* const* d_in, const int* in_sizes, int n_in,
                              void* d_out, int out_size, void* d_ws, size_t ws_size,
                              hipStream_t stream)
{
  (void)in_sizes; (void)n_in; (void)out_size; (void)ws_size;
  const float* F0  = (const float*)d_in[0];   // 3000 x 512
  const float* F1  = (const float*)d_in[1];   // 1500 x 256
  const float* F2  = (const float*)d_in[2];   // 1500 x 256
  const float* FT  = (const float*)d_in[3];   // 3000 x 512
  const float* ADJ = (const float*)d_in[4];   // 5400 x 5400
  const float* MP0 = (const float*)d_in[5];   // 3000 x 3000
  const float* MP1 = (const float*)d_in[6];   // 3000 x 3000
  const unsigned char* MASK = (const unsigned char*)d_in[7]; // 3000 x 3000 bool
  const int* IDX_REAL = (const int*)d_in[8];  // 5400
  const int* IDX_TIR  = (const int*)d_in[9];  // 2400
  const int* IDX_RIT  = (const int*)d_in[10]; // 2400
  const int* IDX_FAKE = (const int*)d_in[11]; // 600
  auto P = [&](int i) { return (const float*)d_in[i]; };
  // params (recursive dict insertion order):
  // 12..17 fc{W,b}x3 | 18..41 full_heads[4]{W,b,a1,a1b,a2,a2b} | 42..47 full_out
  // 48..95 meta_heads[2][4] | 96..107 meta_out[2] | 108..111 fea2mp[2]{W,b}
  // 112..113 fea2embed | 114 red_w | 115 red_b | 116..117 fc1 | 118..119 fc2

  float* out = (float*)d_out;   // [mp0 9e6][mp1 9e6][out 15000][loss 1]
  float* ws  = (float*)d_ws;

  float*  h      = ws + 0L;        // 6000*64
  float*  hreal  = ws + 384000L;   // 5400*64
  float*  z      = ws + 729600L;   // 3000*64
  float*  fts    = ws + 921600L;   // 5400*64 fp32
  __bf16* ftsT   = (__bf16*)(ws + 1267200L); // 64*5400 bf16
  float*  f1b    = ws + 1440000L;  // 5400
  float*  f2b    = ws + 1445440L;  // 5400
  float*  h1cat  = ws + 1450880L;  // 5400*256
  float*  h1     = ws + 2833280L;  // 5400*64
  float*  h1pred = ws + 3178880L;  // 3000*64
  float*  h1full = ws + 3370880L;  // 3000*64
  float*  scat   = ws + 3562880L;  // 3000*256
  float*  sup0   = ws + 4330880L;  // 3000*64
  float*  sup1   = ws + 4522880L;  // 3000*64
  float*  fin    = ws + 4714880L;  // 3000*64
  float*  obuf   = ws + 4906880L;  // 3000*45
  float*  part   = ws + 5041920L;  // 256

  // act=1 (elu), C only
  auto gemm_elu = [&](const float* A, int lda, int M, const float* W, int ldw, int Nc,
                      const float* bias, int K, float* C, int ldc) {
    int tm = (M + 15) / 16, tn = (Nc + 15) / 16, tot = tm * tn;
    gemm_bias_act<1, false, true><<<(tot + 3) / 4, 128, 0, stream>>>(
        A, lda, M, W, ldw, Nc, bias, K, C, ldc, nullptr, 0, tn, tot);
  };
  // act=2 (sigmoid), no bias, C only
  auto gemm_sig = [&](const float* A, int lda, int M, const float* W, int ldw, int Nc,
                      int K, float* C, int ldc) {
    int tm = (M + 15) / 16, tn = (Nc + 15) / 16, tot = tm * tn;
    gemm_bias_act<2, false, false><<<(tot + 3) / 4, 128, 0, stream>>>(
        A, lda, M, W, ldw, Nc, nullptr, K, C, ldc, nullptr, 0, tn, tot);
  };
  // act=0, C + bf16 transposed CT
  auto gemm_ct = [&](const float* A, int lda, int M, const float* W, int ldw, int Nc,
                     const float* bias, int K, float* C, int ldc, __bf16* CT, int ldct) {
    int tm = (M + 15) / 16, tn = (Nc + 15) / 16, tot = tm * tn;
    gemm_bias_act<0, true, true><<<(tot + 3) / 4, 128, 0, stream>>>(
        A, lda, M, W, ldw, Nc, bias, K, C, ldc, CT, ldct, tn, tot);
  };

  // --- input FC layers -> h (6000 x 64), elu ---
  gemm_elu(F0, 512, 3000, P(12), 512, 64, P(13), 512, h,            64);
  gemm_elu(F1, 256, 1500, P(14), 256, 64, P(15), 256, h + 3000L*64, 64);
  gemm_elu(F2, 256, 1500, P(16), 256, 64, P(17), 256, h + 4500L*64, 64);

  // --- fea2mp: z = elu(FT @ W^T + b); mp_pred = sigmoid(z @ z^T) -> d_out ---
  for (int i = 0; i < 2; ++i) {
    gemm_elu(FT, 512, 3000, P(108 + 2*i), 512, 64, P(109 + 2*i), 512, z, 64);
    gemm_sig(z, 64, 3000, z, 64, 3000, 64, out + (long)i * 9000000L, 3000);
  }

  // --- h_real = h[idx_real] ---
  gather64<<<(5400 * 64 + 255) / 256, 256, 0, stream>>>(h, IDX_REAL, hreal, 5400);

  // --- full-graph attention heads (hard mask on adj) -> h1cat (5400 x 256) ---
  for (int hh = 0; hh < 4; ++hh) {
    int bse = 18 + 6 * hh;
    gemm_ct(hreal, 64, 5400, P(bse), 64, 64, P(bse + 1), 64, fts, 64, ftsT, 5400);
    f1f2_kernel<<<(5400 + 255) / 256, 256, 0, stream>>>(fts, P(bse+2), P(bse+3), P(bse+4), P(bse+5), f1b, f2b, 5400);
    attn_fused<false><<<(5400 + 15) / 16, 128, 0, stream>>>(f1b, f2b, ADJ, nullptr, nullptr, nullptr,
                                                            ftsT, 5400, 5400, h1cat, 256, 64 * hh);
  }
  // --- full_out attention -> h1 (5400 x 64) ---
  gemm_ct(h1cat, 256, 5400, P(42), 256, 64, P(43), 256, fts, 64, ftsT, 5400);
  f1f2_kernel<<<(5400 + 255) / 256, 256, 0, stream>>>(fts, P(44), P(45), P(46), P(47), f1b, f2b, 5400);
  attn_fused<false><<<(5400 + 15) / 16, 128, 0, stream>>>(f1b, f2b, ADJ, nullptr, nullptr, nullptr,
                                                          ftsT, 5400, 5400, h1, 64, 0);

  // --- h1_pred = elu(h_target @ fea2embed) ; embed loss ---
  gemm_elu(h, 64, 3000, P(112), 64, 64, P(113), 64, h1pred, 64);
  loss_partial<<<150, 256, 0, stream>>>(h1, h1pred, IDX_TIR, IDX_RIT, part, 2400 * 64);
  loss_finish<<<1, 256, 0, stream>>>(part, 150, out + 18015000L, 1.0f / (2400.0f * 64.0f));

  // --- h1_full = [h1[idx_tir] ; h1_pred[idx_fake]] ---
  gather64<<<(2400 * 64 + 255) / 256, 256, 0, stream>>>(h1, IDX_TIR, h1full, 2400);
  gather64<<<(600 * 64 + 255) / 256, 256, 0, stream>>>(h1pred, IDX_FAKE, h1full + 2400L * 64, 600);

  // --- meta-path attention (soft: logits + where(mask, pred, real_mp)) ---
  for (int i = 0; i < 2; ++i) {
    const float* realm = (i == 0) ? MP0 : MP1;
    const float* pred  = out + (long)i * 9000000L;
    for (int hh = 0; hh < 4; ++hh) {
      int bse = 48 + 24 * i + 6 * hh;
      gemm_ct(h, 64, 3000, P(bse), 64, 64, P(bse + 1), 64, fts, 64, ftsT, 3000);
      f1f2_kernel<<<(3000 + 255) / 256, 256, 0, stream>>>(fts, P(bse+2), P(bse+3), P(bse+4), P(bse+5), f1b, f2b, 3000);
      attn_fused<true><<<(3000 + 15) / 16, 128, 0, stream>>>(f1b, f2b, nullptr, MASK, pred, realm,
                                                             ftsT, 3000, 3000, scat, 256, 64 * hh);
    }
    int ob = 96 + 6 * i;
    gemm_ct(scat, 256, 3000, P(ob), 256, 64, P(ob + 1), 256, fts, 64, ftsT, 3000);
    f1f2_kernel<<<(3000 + 255) / 256, 256, 0, stream>>>(fts, P(ob+2), P(ob+3), P(ob+4), P(ob+5), f1b, f2b, 3000);
    attn_fused<true><<<(3000 + 15) / 16, 128, 0, stream>>>(f1b, f2b, nullptr, MASK, pred, realm,
                                                           ftsT, 3000, 3000, (i == 0 ? sup0 : sup1), 64, 0);
  }

  // --- reduce over {h1_full, sup0, sup1}, classifier head ---
  combine3<<<(3000 * 64 + 255) / 256, 256, 0, stream>>>(h1full, sup0, sup1, P(114), P(115), fin, 3000 * 64);
  gemm_elu(fin, 64, 3000, P(116), 64, 45, P(117), 64, obuf, 45);
  fc2_logsoftmax<<<(3000 + 255) / 256, 256, 0, stream>>>(obuf, P(118), P(119), out + 18000000L, 3000);
}
